// NetTransform_39496519254209
// MI455X (gfx1250) — matrix operand order repaired and checked by reading, not compiled, not run on hardware
//
#include <hip/hip_runtime.h>
#include <math.h>
#include <stdint.h>

typedef _Float16 v16h __attribute__((ext_vector_type(16)));
typedef float    v8f  __attribute__((ext_vector_type(8)));

#define BSZ  4
#define TT   256
#define ID   80      // IDIM == ODIM
#define HD   512
#define TN   10
#define NROW (BSZ*TT)
#define WSWZ (16*5*32*16)   // 40960 = swizzled W1 halves per variant (== 512*80)

union U16h { v16h v; _Float16 h[16]; };
union U8f  { v8f  v; float f[8]; };

__device__ __forceinline__ float sgnf(float x){ return x>0.f?1.f:(x<0.f?-1.f:0.f); }

// ---------------------------------------------------------------- K0: weight prep
// Pre-convert decoder weights to hi/lo f16. W1 stored in WMMA A-fragment order:
// flat = ((kk*5 + m)*32 + lane)*16 + j  ->  W1[m*16 + lane%16][kk*32 + K(lane/16, j)]
// W2 stored in WMMA B-fragment order: flat (kk*32+lane)*80 + n*16 + j -> W2[kk*32 + K(lane/16, j)][n*16 + lane%16].
__global__ __launch_bounds__(256) void prep_w_kernel(
    const float* __restrict__ dlW1, const float* __restrict__ dlW2,
    const float* __restrict__ dhW1, const float* __restrict__ dhW2,
    _Float16* __restrict__ aWh, _Float16* __restrict__ aWl,
    _Float16* __restrict__ w2h, _Float16* __restrict__ w2l)
{
  int v = blockIdx.y;
  int f = blockIdx.x*256 + threadIdx.x;          // 0..40959
  const float* W1 = v? dhW1:dlW1;
  const float* W2 = v? dhW2:dlW2;
  int j   = f & 15;
  int ln  = (f>>4) & 31;
  int mm5 = f >> 9;
  int m   = mm5 % 5, kk = mm5 / 5;
  int grp = ln>>4, lm = ln&15;
  int vv=j>>1, h0=j&1;
  int kl = (vv<4) ? (grp*8 + 2*vv + h0) : (16 + grp*8 + 2*(vv-4) + h0);
  float wa = W1[(m*16+lm)*HD + kk*32 + kl];
  _Float16 ha=(_Float16)wa;
  aWh[v*WSWZ+f]=ha; aWl[v*WSWZ+f]=(_Float16)(wa-(float)ha);
  int r2 = f/ID, c2 = f%ID; int lb = r2&31, jb = c2&15;
  int kb = (jb<8) ? ((lb>>4)*8 + jb) : (16 + (lb>>4)*8 + (jb-8));
  float wb = W2[((r2>>5)*32 + kb)*ID + (c2>>4)*16 + (lb&15)];   // same flat count 512*80
  _Float16 hb=(_Float16)wb;
  w2h[v*WSWZ+f]=hb; w2l[v*WSWZ+f]=(_Float16)(wb-(float)hb);
}

// ---------------------------------------------------------------- K1: encoder
__global__ __launch_bounds__(256) void enc_kernel(
    const float* __restrict__ xin,
    const float* __restrict__ W1, const float* __restrict__ b1,
    const float* __restrict__ W2, const float* __restrict__ b2,
    float* __restrict__ hh, float* __restrict__ hl)
{
  __shared__ float xs[ID];
  __shared__ float a1[HD];
  __shared__ float h2[2*ID];
  __shared__ float st[4];
  int row = blockIdx.x, tid = threadIdx.x;
  if (tid < ID) xs[tid] = xin[row*ID + tid];
  __syncthreads();
  if (tid == 0) {
    float m=0.f; for (int i=0;i<ID;++i) m+=xs[i]; m /= (float)ID;
    float v=0.f; for (int i=0;i<ID;++i){float d=xs[i]-m; v+=d*d;} v /= (float)ID;
    st[0]=m; st[1]=v;
  }
  for (int j=tid;j<HD;j+=256){
    float s=b1[j];
    for (int i=0;i<ID;++i) s += xs[i]*W1[i*HD+j];
    a1[j] = s>0.f ? s : 0.f;
  }
  __syncthreads();
  for (int k=tid;k<2*ID;k+=256){
    float s=b2[k];
    for (int j=0;j<HD;++j) s += a1[j]*W2[j*(2*ID)+k];
    h2[k]=s;
  }
  __syncthreads();
  if (tid==0){
    float m=0.f; for (int k=0;k<2*ID;++k) m+=h2[k]; m /= (float)(2*ID);
    float v=0.f; for (int k=0;k<2*ID;++k){float d=h2[k]-m; v+=d*d;} v /= (float)(2*ID);
    st[2]=m; st[3]=v;
  }
  __syncthreads();
  float inm=st[0], inv=st[1], m2=st[2], v2=st[3];
  if (tid<ID){
    hh[row*ID+tid] = (h2[tid]   -m2)/v2*inv + inm;
    hl[row*ID+tid] = (h2[ID+tid]-m2)/v2*inv + inm;
  }
}

// ---------------------------------------------------------------- K2: decoders
__global__ __launch_bounds__(256) void dec_kernel(
    const float* __restrict__ hl, const float* __restrict__ hh,
    const float* __restrict__ xout,                       // (B,T,TN,ID)
    const float* __restrict__ dlW1,const float* __restrict__ dlb1,
    const float* __restrict__ dlW2,const float* __restrict__ dlb2,
    const float* __restrict__ dhW1,const float* __restrict__ dhb1,
    const float* __restrict__ dhW2,const float* __restrict__ dhb2,
    float* __restrict__ out_l, float* __restrict__ out_h,
    float* __restrict__ bhat_l, float* __restrict__ bhat_h,
    unsigned* __restrict__ rm_l, unsigned* __restrict__ rm_h)
{
  int row=blockIdx.x, var=blockIdx.y, tid=threadIdx.x;
  const float* W1 = var? dhW1:dlW1; const float* b1 = var? dhb1:dlb1;
  const float* W2 = var? dhW2:dlW2; const float* b2 = var? dhb2:dlb2;
  const float* srow = (var? hh:hl) + row*ID;
  __shared__ float s[ID], zr[HD], b1m[HD], outp[ID], bh[ID], st[4];
  __shared__ unsigned rm[16];
  if (tid<ID) s[tid]=srow[tid];
  if (tid<16) rm[tid]=0u;
  __syncthreads();
  for (int j=tid;j<HD;j+=256){
    float zz=b1[j];
    for (int i=0;i<ID;++i) zz += s[i]*W1[i*HD+j];
    bool on = zz>0.f;
    zr[j]  = on ? zz    : 0.f;      // relu(z)
    b1m[j] = on ? b1[j] : 0.f;      // rr * b1
    if (on) atomicOr(&rm[j>>5], 1u<<(j&31));
  }
  __syncthreads();
  for (int k=tid;k<ID;k+=256){
    float so=b2[k], sb=b2[k];
    for (int j=0;j<HD;++j){
      float w=W2[j*ID+k];
      so += zr[j]*w;
      sb += b1m[j]*w;
    }
    outp[k]=so; bh[k]=sb;
  }
  __syncthreads();
  if (tid==0){
    float m=0.f; for (int k=0;k<ID;++k) m+=outp[k]; m /= (float)ID;
    float v=0.f; for (int k=0;k<ID;++k){float d=outp[k]-m; v+=d*d;} v /= (float)ID;
    st[0]=m; st[1]=v;
    const float* xr = xout + (size_t)(row*TN+0)*ID;      // n = 0 stats only
    float om=0.f; for (int k=0;k<ID;++k) om+=xr[k]; om /= (float)ID;
    float ov=0.f; for (int k=0;k<ID;++k){float d=xr[k]-om; ov+=d*d;} ov /= (float)ID;
    st[2]=om; st[3]=ov;
  }
  __syncthreads();
  if (tid<ID){
    float val=(outp[tid]-st[0])/st[1]*st[3]+st[2];
    (var? out_h :out_l )[row*ID+tid]=val;
    (var? bhat_h:bhat_l)[row*ID+tid]=bh[tid];
  }
  if (tid<16) (var? rm_h:rm_l)[row*16+tid]=rm[tid];
}

// ---------------------------------------------------------------- K3: energy (WMMA)
// w_hat = (W1 masked by relu-mask) @ W2 per (b,t) row; A/B fragments loaded
// directly from the pre-swizzled f16 hi/lo arrays; mask applied via cndmask.
__global__ __launch_bounds__(160) void energy_kernel(
    const float* __restrict__ xout,
    const float* __restrict__ hl, const float* __restrict__ hh,
    const float* __restrict__ bhat_l, const float* __restrict__ bhat_h,
    const unsigned* __restrict__ rm_l, const unsigned* __restrict__ rm_h,
    const _Float16* __restrict__ aWh, const _Float16* __restrict__ aWl,
    const _Float16* __restrict__ w2h, const _Float16* __restrict__ w2l,
    float* __restrict__ e0_l, float* __restrict__ e0_h)
{
  int row=blockIdx.x, var=blockIdx.y, tid=threadIdx.x;
  const float* srow = (var? hh:hl) + row*ID;
  const float* bhp  = (var? bhat_h:bhat_l) + row*ID;
  const unsigned* rm= (var? rm_h:rm_l) + row*16;
  const _Float16* Ah = aWh + var*WSWZ;
  const _Float16* Al = aWl + var*WSWZ;
  const _Float16* Bh = w2h + var*WSWZ;
  const _Float16* Bl = w2l + var*WSWZ;
  float* eo = var? e0_h : e0_l;

  __shared__ float Wm[80*81];
  __shared__ float ssg[80], sab[80], esg[80];
  __shared__ unsigned rb[16];
  __shared__ float rnum[80], rden[80];

  if (tid<16) rb[tid]=rm[tid];
  if (tid<80){
    float sv=srow[tid]; ssg[tid]=sgnf(sv); sab[tid]=fabsf(sv);
    float ev = xout[(size_t)(row*TN+0)*ID + tid] - bhp[tid];   // n = 0 only
    esg[tid]=sgnf(ev);
  }
  __syncthreads();

  int wv=tid>>5, ln=tid&31, grp=ln>>4, lm=ln&15;
  U8f acc[5];
  #pragma unroll
  for (int n=0;n<5;++n)
    #pragma unroll
    for (int r=0;r<8;++r) acc[n].f[r]=0.f;

  for (int kk=0; kk<16; ++kk){
    // A fragment (hi/lo) for this wave's M-tile, direct 32B loads
    U16h awh, awl, afh, afl;
    size_t aoff = (size_t)((kk*5 + wv)*32 + ln)*16;
    awh.v = *(const v16h*)(Ah + aoff);
    awl.v = *(const v16h*)(Al + aoff);
    // mask bits for this lane's 16 fragment elements = 2 bytes of the 512b mask
    unsigned word = rb[kk];
    unsigned m0 = (word >> (grp*8))      & 0xffu;
    unsigned m1 = (word >> (16 + grp*8)) & 0xffu;
    #pragma unroll
    for (int j=0;j<16;++j){
      bool on = (j<8) ? ((m0>>j)&1u) : ((m1>>(j-8))&1u);
      afh.h[j] = on ? awh.h[j] : (_Float16)0;
      afl.h[j] = on ? awl.h[j] : (_Float16)0;
    }
    size_t boff = (size_t)(kk*32 + ln)*ID;
    #pragma unroll
    for (int n=0;n<5;++n){
      v16h bfh = *(const v16h*)(Bh + boff + n*16);
      v16h bfl = *(const v16h*)(Bl + boff + n*16);
      acc[n].v = __builtin_amdgcn_wmma_f32_16x16x32_f16(false,afh.v,false,bfh,(short)0,acc[n].v,false,false);
      acc[n].v = __builtin_amdgcn_wmma_f32_16x16x32_f16(false,afh.v,false,bfl,(short)0,acc[n].v,false,false);
      acc[n].v = __builtin_amdgcn_wmma_f32_16x16x32_f16(false,afl.v,false,bfh,(short)0,acc[n].v,false,false);
    }
  }
  #pragma unroll
  for (int n=0;n<5;++n)
    #pragma unroll
    for (int r=0;r<8;++r){
      int M = wv*16 + ((ln<16)? r : r+8);
      int N = n*16 + lm;
      Wm[M*81+N]=acc[n].f[r];
    }
  __syncthreads();
  if (tid<80){
    float p=0.f, nn=0.f, si=ssg[tid];
    for (int o=0;o<80;++o){
      float w = Wm[tid*81+o]*si*esg[o];
      p += fmaxf(w,0.f); nn += fmaxf(-w,0.f);
    }
    rnum[tid]=sab[tid]*nn; rden[tid]=sab[tid]*p;
  }
  __syncthreads();
  if (tid==0){
    float num=0.f, den=0.f;
    for (int i=0;i<80;++i){ num+=rnum[i]; den+=rden[i]; }
    float en = num/den;
    if (en != en) en = 1.f;                 // NaN -> 1.0
    eo[row] = fmaxf(1.f-en, 0.f);
  }
}

// ---------------------------------------------------------------- K4: min/max
__global__ __launch_bounds__(1024) void minmax_kernel(
    const float* __restrict__ e0l, const float* __restrict__ e0h, float* __restrict__ mm)
{
  __shared__ float smn[1024], smx[1024];
  int tid=threadIdx.x;
  for (int v=0; v<2; ++v){
    const float* e = v? e0h:e0l;
    float x=e[tid]; smn[tid]=x; smx[tid]=x;
    __syncthreads();
    for (int s=512;s>0;s>>=1){
      if (tid<s){ smn[tid]=fminf(smn[tid],smn[tid+s]); smx[tid]=fmaxf(smx[tid],smx[tid+s]); }
      __syncthreads();
    }
    if (tid==0){ mm[v*2]=smn[0]; mm[v*2+1]=smx[0]; }
    __syncthreads();
  }
}

// ---------------------------------------------------------------- K5: _fb
__global__ __launch_bounds__(256) void fb_kernel(
    const float* __restrict__ e0, const float* __restrict__ mm, int vsel,
    float* __restrict__ out)
{
  __shared__ float xn[TT];
  int b=blockIdx.x, j=threadIdx.x;
  float mn=mm[vsel*2], mx=mm[vsel*2+1];
  xn[j] = 1.f - (e0[b*TT+j]-mn)/(mx-mn);
  __syncthreads();
  float* o = out + (size_t)b*TT*TT;
  float p=1.f;
  for (int i=0;i<TT;++i){                         // forward cumprod (axis 0)
    float val = (i<j)?1.f : ((i==j)? xn[i]+1.f : xn[i]);
    p *= val;
    float mf = (i<=j)?1.f:0.f;
    o[i*TT+j] = 0.5f*(p-mf);
  }
  float q=1.f;
  for (int i=TT-1;i>=0;--i){                      // backward cumprod
    float val = (i>j)?1.f : ((i==j)? xn[i]+1.f : xn[i]);
    q *= val;
    float mb = (i>=j)?1.f:0.f;
    o[i*TT+j] += 0.5f*(q-mb);
  }
}

// ---------------------------------------------------------------- K6: gram + sigmoid (WMMA)
__global__ __launch_bounds__(256) void gram_kernel(
    const float* __restrict__ hh, float* __restrict__ out)
{
  __shared__ _Float16 aH[8][512], aL[8][512], bH[8][512], bL[8][512];
  int tid=threadIdx.x, wid=tid>>5, ln=tid&31, grp=ln>>4, lm=ln&15;
  int wg = blockIdx.x*8 + wid;
  int b = wg>>8, tile = wg&255;
  int tm = tile>>4, tn = tile&15;
  U8f acc;
  #pragma unroll
  for (int r=0;r<8;++r) acc.f[r]=0.f;

  for (int c=0;c<3;++c){
    int kk=c*32;
    for (int q=0;q<16;++q){
      int f=q*32+ln;                     // 0..511
      int r=f>>5, col=f&31;
      float va = (kk+col<ID)? hh[(size_t)(b*TT + tm*16 + r)*ID + kk+col] : 0.f;
      _Float16 ha=(_Float16)va; aH[wid][f]=ha; aL[wid][f]=(_Float16)(va-(float)ha);
      float vb = (kk+col<ID)? hh[(size_t)(b*TT + tn*16 + r)*ID + kk+col] : 0.f;
      _Float16 hb=(_Float16)vb; bH[wid][f]=hb; bL[wid][f]=(_Float16)(vb-(float)hb);
    }
    __syncthreads();
    U16h afh, afl, bfh, bfl;
    #pragma unroll
    for (int j=0;j<16;++j){
      int vv=j>>1, h0=j&1;
      int kl = (vv<4) ? (grp*8 + 2*vv + h0) : (16 + grp*8 + 2*(vv-4) + h0);
      int ia = lm*32 + kl;               // A: M=lm row, K=kl
      afh.h[j]=aH[wid][ia]; afl.h[j]=aL[wid][ia];
      int ib = lm*32 + kl;               // B[k=kl][n=lm] = H[tn*16+lm][kk+kl]
      bfh.h[j]=bH[wid][ib]; bfl.h[j]=bL[wid][ib];
    }
    acc.v = __builtin_amdgcn_wmma_f32_16x16x32_f16(false,afh.v,false,bfh.v,(short)0,acc.v,false,false);
    acc.v = __builtin_amdgcn_wmma_f32_16x16x32_f16(false,afh.v,false,bfl.v,(short)0,acc.v,false,false);
    acc.v = __builtin_amdgcn_wmma_f32_16x16x32_f16(false,afl.v,false,bfh.v,(short)0,acc.v,false,false);
    __syncthreads();
  }
  #pragma unroll
  for (int r=0;r<8;++r){
    int M = tm*16 + ((ln<16)? r : r+8);
    int N = tn*16 + lm;
    float g=acc.f[r];
    out[(size_t)b*TT*TT + M*TT + N] = 1.f/(1.f+expf(-g));
  }
}

// ---------------------------------------------------------------- K7: loss
__global__ __launch_bounds__(256) void loss_partial_kernel(
    const float* __restrict__ out_l, const float* __restrict__ out_h,
    const float* __restrict__ xout, const float* __restrict__ kth,
    const int* __restrict__ ilens, float* __restrict__ partial)
{
  int row=blockIdx.x, b=row>>8, t=row&255, tid=threadIdx.x;
  __shared__ float sb[256];
  float acc=0.f;
  if (t < ilens[b]){
    for (int idx=tid; idx<TN*ID; idx+=256){
      int n=idx/ID, i=idx-n*ID;
      float d = out_l[row*ID+i] - xout[(size_t)(row*TN+n)*ID+i];
      float kw = (t+n<TT)? kth[(size_t)b*TT*TT + t*TT + (t+n)] : 0.f;
      acc += d*d*kw;
    }
    for (int i=tid;i<ID;i+=256){
      float d = out_h[row*ID+i] - xout[(size_t)(row*TN+0)*ID+i];
      acc += d*d;
    }
  }
  sb[tid]=acc; __syncthreads();
  for (int s=128;s>0;s>>=1){ if (tid<s) sb[tid]+=sb[tid+s]; __syncthreads(); }
  if (tid==0) partial[row]=sb[0];
}

__global__ __launch_bounds__(256) void loss_final_kernel(
    const float* __restrict__ partial, float* __restrict__ out)
{
  __shared__ float sb[256];
  int tid=threadIdx.x;
  float s=0.f;
  for (int k=tid;k<NROW;k+=256) s+=partial[k];
  sb[tid]=s; __syncthreads();
  for (int st=128;st>0;st>>=1){ if (tid<st) sb[tid]+=sb[tid+st]; __syncthreads(); }
  if (tid==0) out[0]=sb[0];
}

// ---------------------------------------------------------------- launch
extern "C" void kernel_launch(void* const* d_in, const int* in_sizes, int n_in,
                              void* d_out, int out_size, void* d_ws, size_t ws_size,
                              hipStream_t stream) {
  const float* xin   = (const float*)d_in[0];
  const float* xout  = (const float*)d_in[1];
  const int*   ilens = (const int*)  d_in[2];
  const float* encW1 = (const float*)d_in[4];
  const float* encb1 = (const float*)d_in[5];
  const float* encW2 = (const float*)d_in[6];
  const float* encb2 = (const float*)d_in[7];
  const float* dlW1  = (const float*)d_in[8];
  const float* dlb1  = (const float*)d_in[9];
  const float* dlW2  = (const float*)d_in[10];
  const float* dlb2  = (const float*)d_in[11];
  const float* dhW1  = (const float*)d_in[12];
  const float* dhb1  = (const float*)d_in[13];
  const float* dhW2  = (const float*)d_in[14];
  const float* dhb2  = (const float*)d_in[15];
  float* fout = (float*)d_out;     // [0]=loss, [1..262144]=kernel, then kernel_target_l

  float* ws     = (float*)d_ws;
  float* hh     = ws;                 // 81920
  float* hl     = hh + NROW*ID;
  float* out_l  = hl + NROW*ID;
  float* out_h  = out_l + NROW*ID;
  float* bhat_l = out_h + NROW*ID;
  float* bhat_h = bhat_l + NROW*ID;
  unsigned* rm_l = (unsigned*)(bhat_h + NROW*ID);
  unsigned* rm_h = rm_l + NROW*16;
  float* e0_l   = (float*)(rm_h + NROW*16);
  float* e0_h   = e0_l + NROW;
  float* mm     = e0_h + NROW;
  float* kth    = mm + 4;             // 262144
  float* partial= kth + BSZ*TT*TT;    // 1024
  // 32B-aligned f16 weight arrays (for v16h global loads)
  uintptr_t p = (uintptr_t)(partial + NROW);
  p = (p + 31u) & ~(uintptr_t)31u;
  _Float16* aWh = (_Float16*)p;       // 2*WSWZ halves each
  _Float16* aWl = aWh + 2*WSWZ;
  _Float16* w2h = aWl + 2*WSWZ;
  _Float16* w2l = w2h + 2*WSWZ;

  prep_w_kernel<<<dim3(WSWZ/256,2),256,0,stream>>>(dlW1,dlW2,dhW1,dhW2,aWh,aWl,w2h,w2l);
  enc_kernel<<<NROW,256,0,stream>>>(xin, encW1,encb1,encW2,encb2, hh, hl);
  dec_kernel<<<dim3(NROW,2),256,0,stream>>>(hl,hh,xout,
      dlW1,dlb1,dlW2,dlb2, dhW1,dhb1,dhW2,dhb2,
      out_l,out_h,bhat_l,bhat_h,rm_l,rm_h);
  energy_kernel<<<dim3(NROW,2),160,0,stream>>>(xout,hl,hh,bhat_l,bhat_h,
      rm_l,rm_h,aWh,aWl,w2h,w2l,e0_l,e0_h);
  minmax_kernel<<<1,1024,0,stream>>>(e0_l,e0_h,mm);
  fb_kernel<<<BSZ,256,0,stream>>>(e0_l, mm, 0, fout + 1 + BSZ*TT*TT); // kernel_target_l
  fb_kernel<<<BSZ,256,0,stream>>>(e0_h, mm, 1, kth);                  // kernel_target_h
  gram_kernel<<<128,256,0,stream>>>(hh, fout + 1);                    // kernel
  loss_partial_kernel<<<NROW,256,0,stream>>>(out_l,out_h,xout,kth,ilens,partial);
  loss_final_kernel<<<1,256,0,stream>>>(partial, fout);
}
